// DecoderSumCover_53171695124927
// MI455X (gfx1250) — matrix operand order repaired and checked by reading, not compiled, not run on hardware
//
#include <hip/hip_runtime.h>
#include <hip/hip_bf16.h>
#include <math.h>

// Problem constants (match reference)
#define HDIM 512
#define VDIM 4000
#define MDIM 50
#define HOPS 6

typedef __bf16 bf16_t;
typedef bf16_t bf16x8 __attribute__((ext_vector_type(8)));
typedef bf16_t v16bf  __attribute__((ext_vector_type(16)));
typedef float  v8f    __attribute__((ext_vector_type(8)));

static __device__ __forceinline__ bf16_t to_bf16(float f) {
    // round-to-nearest-even f32 -> bf16
    unsigned int u = __builtin_bit_cast(unsigned int, f);
    unsigned int r = u + 0x7FFFu + ((u >> 16) & 1u);
    unsigned short s = (unsigned short)(r >> 16);
    return __builtin_bit_cast(bf16_t, s);
}

// ---------------------------------------------------------------------------
// bf16 WMMA GEMM: C[M,N] = A[M,K] @ B[K,N], with B given pre-transposed as
// Bt[N,K] so every lane's B fragment is two runs of 8 contiguous bf16 values.
// Each wave computes a 32x32 tile (2x2 of 16x16 accumulators).
// MODE 0: f32 row-major out, + bias[n]            (emb = adj@T2 + b2)
// MODE 1: bf16 row-major out, relu(x + bias[n])   (h   = relu(adj@W1 + b1))
// MODE 2: bf16 transposed out [N,M], no bias      (T2t = (h@W2)^T)
// ---------------------------------------------------------------------------
template <int MODE>
__global__ __launch_bounds__(256)
void gemm_bf16_wmma(const bf16_t* __restrict__ A, const bf16_t* __restrict__ Bt,
                    void* __restrict__ outv, const float* __restrict__ bias,
                    int M, int N, int K)
{
    const int wave = (int)((blockIdx.x * blockDim.x + threadIdx.x) >> 5);
    const int lane = (int)(threadIdx.x & 31);
    const int half = lane >> 4;   // lane half selects K sub-range per ISA layout
    const int lr   = lane & 15;   // row (A) / col (B) within tile

    const int tilesN = N >> 5;
    const int tm = (wave / tilesN) << 5;
    const int tn = (wave % tilesN) << 5;
    if (tm >= M) return;

    v8f c00 = {}, c01 = {}, c10 = {}, c11 = {};

    // A 16x32 bf16 fragment (per lane): rows tm+lr / tm+16+lr,
    //   chunk0 = K[k0+8*half .. +7], chunk1 = K[k0+16+8*half .. +7]
    const bf16_t* a0p = A + (size_t)(tm + lr) * K + 8 * half;
    const bf16_t* a1p = a0p + (size_t)16 * K;
    // B 32x16 bf16 fragment (per lane, from Bt[N,K]): col tn+lr / tn+16+lr,
    //   chunk0 = K[k0+8*half .. +7], chunk1 = K[k0+16+8*half .. +7]
    const bf16_t* b0p = Bt + (size_t)(tn + lr) * K + 8 * half;
    const bf16_t* b1p = b0p + (size_t)16 * K;

    for (int k0 = 0; k0 < K; k0 += 32) {
        bf16x8 a0lo = *(const bf16x8*)(a0p + k0);
        bf16x8 a0hi = *(const bf16x8*)(a0p + k0 + 16);
        bf16x8 a1lo = *(const bf16x8*)(a1p + k0);
        bf16x8 a1hi = *(const bf16x8*)(a1p + k0 + 16);
        bf16x8 b0lo = *(const bf16x8*)(b0p + k0);
        bf16x8 b0hi = *(const bf16x8*)(b0p + k0 + 16);
        bf16x8 b1lo = *(const bf16x8*)(b1p + k0);
        bf16x8 b1hi = *(const bf16x8*)(b1p + k0 + 16);

        v16bf a0, a1, b0, b1;
#pragma unroll
        for (int i = 0; i < 8; ++i) {
            a0[i] = a0lo[i]; a0[8 + i] = a0hi[i];
            a1[i] = a1lo[i]; a1[8 + i] = a1hi[i];
            b0[i] = b0lo[i]; b0[8 + i] = b0hi[i];
            b1[i] = b1lo[i]; b1[8 + i] = b1hi[i];
        }

        c00 = __builtin_amdgcn_wmma_f32_16x16x32_bf16(false, a0, false, b0, (short)0, c00, false, false);
        c01 = __builtin_amdgcn_wmma_f32_16x16x32_bf16(false, a0, false, b1, (short)0, c01, false, false);
        c10 = __builtin_amdgcn_wmma_f32_16x16x32_bf16(false, a1, false, b0, (short)0, c10, false, false);
        c11 = __builtin_amdgcn_wmma_f32_16x16x32_bf16(false, a1, false, b1, (short)0, c11, false, false);
    }

    // C/D 16x16 f32 layout: VGPR r -> M = 8*half + r, N = lr
    auto store_tile = [&](const v8f& c, int bm, int bn) {
        const int n = tn + bn + lr;
        float bv = 0.0f;
        if (MODE != 2 && bias != nullptr) bv = bias[n];
#pragma unroll
        for (int r = 0; r < 8; ++r) {
            const int m = tm + bm + 8 * half + r;
            const float val = c[r];
            if (MODE == 0) {
                ((float*)outv)[(size_t)m * N + n] = val + bv;
            } else if (MODE == 1) {
                float x = val + bv;
                ((bf16_t*)outv)[(size_t)m * N + n] = to_bf16(x > 0.0f ? x : 0.0f);
            } else {
                ((bf16_t*)outv)[(size_t)n * M + m] = to_bf16(val);
            }
        }
    };
    store_tile(c00, 0, 0);
    store_tile(c01, 0, 16);
    store_tile(c10, 16, 0);
    store_tile(c11, 16, 16);
}

// ---------------------------------------------------------------------------
// f32 -> bf16 conversions
// ---------------------------------------------------------------------------
__global__ void conv_bf16(const float* __restrict__ in, bf16_t* __restrict__ out, size_t n)
{
    size_t i = (size_t)blockIdx.x * blockDim.x + threadIdx.x;
    size_t stride = (size_t)gridDim.x * blockDim.x;
    for (; i < n; i += stride) out[i] = to_bf16(in[i]);
}

// in [K,N] row-major -> out [N,K] bf16
__global__ void tconv_bf16(const float* __restrict__ in, bf16_t* __restrict__ out, int K, int N)
{
    size_t total = (size_t)K * N;
    size_t i = (size_t)blockIdx.x * blockDim.x + threadIdx.x;
    size_t stride = (size_t)gridDim.x * blockDim.x;
    for (; i < total; i += stride) {
        int n = (int)(i / K);
        int k = (int)(i % K);
        out[i] = to_bf16(in[(size_t)k * N + n]);
    }
}

// ---------------------------------------------------------------------------
// Tiny f32 kernels for the hops / softmaxes / head
// ---------------------------------------------------------------------------
__global__ void k_init_lq(const float* __restrict__ q, float* __restrict__ lq)
{
    int i = blockIdx.x * blockDim.x + threadIdx.x;
    if (i < HDIM) lq[i] = q[i];
}

// aw = softmax_j( dot(lq, keys[j]) ), j < MDIM   (one block of 64 threads)
__global__ void k_attn_mem(const float* __restrict__ lq, const float* __restrict__ keys,
                           float* __restrict__ aw)
{
    __shared__ float s[MDIM];
    int j = threadIdx.x;
    if (j < MDIM) {
        float acc = 0.0f;
        for (int i = 0; i < HDIM; ++i) acc += lq[i] * keys[(size_t)j * HDIM + i];
        s[j] = acc;
    }
    __syncthreads();
    if (j == 0) {
        float mx = -1e30f;
        for (int t = 0; t < MDIM; ++t) mx = fmaxf(mx, s[t]);
        float sm = 0.0f;
        for (int t = 0; t < MDIM; ++t) { float e = __expf(s[t] - mx); aw[t] = e; sm += e; }
        float inv = 1.0f / sm;
        for (int t = 0; t < MDIM; ++t) aw[t] *= inv;
    }
}

// rvec[v] = sum_j aw[j] * mv[j,v]
__global__ void k_aw_mv(const float* __restrict__ aw, const float* __restrict__ mv,
                        float* __restrict__ rvec)
{
    int v = blockIdx.x * blockDim.x + threadIdx.x;
    if (v < VDIM) {
        float acc = 0.0f;
#pragma unroll
        for (int j = 0; j < MDIM; ++j) acc += aw[j] * mv[(size_t)j * VDIM + v];
        rvec[v] = acc;
    }
}

// out[h] (+)= sum_v w[v] * emb[v,h]  — one block per h, 256-thread reduction
template <int ACC>
__global__ void k_vecmat_emb(const float* __restrict__ w, const float* __restrict__ emb,
                             float* __restrict__ out)
{
    __shared__ float red[256];
    const int h = blockIdx.x;
    float acc = 0.0f;
    for (int v = threadIdx.x; v < VDIM; v += blockDim.x)
        acc += w[v] * emb[(size_t)v * HDIM + h];
    red[threadIdx.x] = acc;
    __syncthreads();
    for (int s = blockDim.x >> 1; s > 0; s >>= 1) {
        if ((int)threadIdx.x < s) red[threadIdx.x] += red[threadIdx.x + s];
        __syncthreads();
    }
    if (threadIdx.x == 0) {
        if (ACC) out[h] += red[0]; else out[h] = red[0];
    }
}

// es[v] = dot(query, emb[v,:])  — one block per v (coalesced over H)
__global__ void k_escores(const float* __restrict__ q, const float* __restrict__ emb,
                          float* __restrict__ es)
{
    __shared__ float red[128];
    const int v = blockIdx.x;
    float acc = 0.0f;
    for (int i = threadIdx.x; i < HDIM; i += blockDim.x)
        acc += q[i] * emb[(size_t)v * HDIM + i];
    red[threadIdx.x] = acc;
    __syncthreads();
    for (int s = blockDim.x >> 1; s > 0; s >>= 1) {
        if ((int)threadIdx.x < s) red[threadIdx.x] += red[threadIdx.x + s];
        __syncthreads();
    }
    if (threadIdx.x == 0) es[v] = red[0];
}

// awe = softmax(es) over VDIM  (single 256-thread block)
__global__ void k_softmax_v(const float* __restrict__ es, float* __restrict__ awe)
{
    __shared__ float red[256];
    float mx = -1e30f;
    for (int v = threadIdx.x; v < VDIM; v += 256) mx = fmaxf(mx, es[v]);
    red[threadIdx.x] = mx;
    __syncthreads();
    for (int s = 128; s > 0; s >>= 1) {
        if ((int)threadIdx.x < s) red[threadIdx.x] = fmaxf(red[threadIdx.x], red[threadIdx.x + s]);
        __syncthreads();
    }
    mx = red[0];
    __syncthreads();
    float sm = 0.0f;
    for (int v = threadIdx.x; v < VDIM; v += 256) sm += __expf(es[v] - mx);
    red[threadIdx.x] = sm;
    __syncthreads();
    for (int s = 128; s > 0; s >>= 1) {
        if ((int)threadIdx.x < s) red[threadIdx.x] += red[threadIdx.x + s];
        __syncthreads();
    }
    const float inv = 1.0f / red[0];
    for (int v = threadIdx.x; v < VDIM; v += 256) awe[v] = __expf(es[v] - mx) * inv;
}

// hidden = relu( relu(concat(q, lq, ce)) @ Wo1 + bo1 )
__global__ void k_head1(const float* __restrict__ q, const float* __restrict__ lq,
                        const float* __restrict__ ce, const float* __restrict__ Wo1,
                        const float* __restrict__ bo1, float* __restrict__ hidden)
{
    __shared__ float feat[3 * HDIM];
    for (int i = threadIdx.x; i < 3 * HDIM; i += blockDim.x) {
        float x = (i < HDIM) ? q[i] : (i < 2 * HDIM ? lq[i - HDIM] : ce[i - 2 * HDIM]);
        feat[i] = x > 0.0f ? x : 0.0f;
    }
    __syncthreads();
    const int n = blockIdx.x * blockDim.x + threadIdx.x;
    if (n < 2 * HDIM) {
        float acc = bo1[n];
        for (int k = 0; k < 3 * HDIM; ++k) acc += feat[k] * Wo1[(size_t)k * (2 * HDIM) + n];
        hidden[n] = acc > 0.0f ? acc : 0.0f;
    }
}

// out = hidden @ Wo2 + bo2
__global__ void k_head2(const float* __restrict__ hidden, const float* __restrict__ Wo2,
                        const float* __restrict__ bo2, float* __restrict__ out)
{
    __shared__ float hsh[2 * HDIM];
    for (int i = threadIdx.x; i < 2 * HDIM; i += blockDim.x) hsh[i] = hidden[i];
    __syncthreads();
    const int v = blockIdx.x * blockDim.x + threadIdx.x;
    if (v < VDIM) {
        float acc = bo2[v];
        for (int k = 0; k < 2 * HDIM; ++k) acc += hsh[k] * Wo2[(size_t)k * VDIM + v];
        out[v] = acc;
    }
}

// ---------------------------------------------------------------------------
// Launcher
// ---------------------------------------------------------------------------
extern "C" void kernel_launch(void* const* d_in, const int* in_sizes, int n_in,
                              void* d_out, int out_size, void* d_ws, size_t ws_size,
                              hipStream_t stream)
{
    (void)in_sizes; (void)n_in; (void)out_size; (void)ws_size;

    const float* query = (const float*)d_in[0];
    const float* keys  = (const float*)d_in[1];
    const float* mv    = (const float*)d_in[2];
    const float* adj   = (const float*)d_in[3];
    const float* W1    = (const float*)d_in[4];
    const float* b1    = (const float*)d_in[5];
    const float* W2    = (const float*)d_in[6];
    const float* b2    = (const float*)d_in[7];
    const float* Wo1   = (const float*)d_in[8];
    const float* bo1   = (const float*)d_in[9];
    const float* Wo2   = (const float*)d_in[10];
    const float* bo2   = (const float*)d_in[11];
    float* out = (float*)d_out;

    // Deterministic workspace carve-out (~53 MB total)
    char* ws = (char*)d_ws;
    size_t cur = 0;
    auto carve = [&](size_t bytes) -> char* {
        char* p = ws + cur;
        cur = (cur + bytes + 255) & ~(size_t)255;
        return p;
    };
    bf16_t* adjb = (bf16_t*)carve((size_t)VDIM * VDIM * sizeof(bf16_t));
    bf16_t* w1t  = (bf16_t*)carve((size_t)HDIM * VDIM * sizeof(bf16_t));
    bf16_t* hbf  = (bf16_t*)carve((size_t)VDIM * HDIM * sizeof(bf16_t));
    bf16_t* w2t  = (bf16_t*)carve((size_t)HDIM * HDIM * sizeof(bf16_t));
    bf16_t* t2t  = (bf16_t*)carve((size_t)HDIM * VDIM * sizeof(bf16_t));
    float*  emb  = (float*)carve((size_t)VDIM * HDIM * sizeof(float));
    float*  lq   = (float*)carve(HDIM * sizeof(float));
    float*  aw   = (float*)carve(MDIM * sizeof(float));
    float*  rvec = (float*)carve(VDIM * sizeof(float));
    float*  es   = (float*)carve(VDIM * sizeof(float));
    float*  awe  = (float*)carve(VDIM * sizeof(float));
    float*  ce   = (float*)carve(HDIM * sizeof(float));
    float*  hid  = (float*)carve(2 * HDIM * sizeof(float));

    // 1) precision conversion for the WMMA GEMMs
    conv_bf16<<<2048, 256, 0, stream>>>(adj, adjb, (size_t)VDIM * VDIM);
    tconv_bf16<<<2048, 256, 0, stream>>>(W1, w1t, VDIM, HDIM);   // W1[V,H]  -> W1t[H,V]
    tconv_bf16<<<512, 256, 0, stream>>>(W2, w2t, HDIM, HDIM);    // W2[H,H]  -> W2t[H,H]

    // 2) GCN via WMMA: waves = (M/32)*(N/32) = 125*16 = 2000 -> 250 blocks of 8 waves
    const int gemm_blocks = ((VDIM / 32) * (HDIM / 32) * 32 + 255) / 256;
    gemm_bf16_wmma<1><<<gemm_blocks, 256, 0, stream>>>(adjb, w1t, hbf, b1, VDIM, HDIM, VDIM);
    gemm_bf16_wmma<2><<<gemm_blocks, 256, 0, stream>>>(hbf, w2t, t2t, nullptr, VDIM, HDIM, HDIM);
    gemm_bf16_wmma<0><<<gemm_blocks, 256, 0, stream>>>(adjb, t2t, emb, b2, VDIM, HDIM, VDIM);

    // 3) multi-hop key-value memory read
    k_init_lq<<<(HDIM + 255) / 256, 256, 0, stream>>>(query, lq);
    for (int hop = 0; hop < HOPS; ++hop) {
        k_attn_mem<<<1, 64, 0, stream>>>(lq, keys, aw);
        k_aw_mv<<<(VDIM + 255) / 256, 256, 0, stream>>>(aw, mv, rvec);
        k_vecmat_emb<1><<<HDIM, 256, 0, stream>>>(rvec, emb, lq);
    }

    // 4) entity attention context (uses original query)
    k_escores<<<VDIM, 128, 0, stream>>>(query, emb, es);
    k_softmax_v<<<1, 256, 0, stream>>>(es, awe);
    k_vecmat_emb<0><<<HDIM, 256, 0, stream>>>(awe, emb, ce);

    // 5) output head
    k_head1<<<(2 * HDIM + 255) / 256, 256, 0, stream>>>(query, lq, ce, Wo1, bo1, hid);
    k_head2<<<(VDIM + 255) / 256, 256, 0, stream>>>(hid, Wo2, bo2, out);
}
